// EmpathyAttention_74938589380613
// MI455X (gfx1250) — matrix operand order repaired and checked by reading, not compiled, not run on hardware
//
#include <hip/hip_runtime.h>

#define BATCH 2
#define SEQ   2048
#define DM    1024
#define NH    16
#define DKH   64
#define MROWS (BATCH * SEQ)   // 4096

typedef __attribute__((ext_vector_type(4)))  __bf16 v4bf;
typedef __attribute__((ext_vector_type(8)))  __bf16 v8bf;
typedef __attribute__((ext_vector_type(16))) __bf16 v16bf;
typedef __attribute__((ext_vector_type(8)))  float  v8f;

static __device__ __forceinline__ __bf16 f2bf(float f) {
    union { float f; unsigned u; } x{f};
    unsigned r = x.u + 0x7FFFu + ((x.u >> 16) & 1u);   // round-to-nearest-even
    unsigned short h = (unsigned short)(r >> 16);
    return __builtin_bit_cast(__bf16, h);
}

// A-matrix fragment (16x32 bf16, MxK): lane (m = lane&15, half = lane>>4)
// a[0..7] = row[8*half .. +7], a[8..15] = row[16+8*half .. +7]
static __device__ __forceinline__ v16bf load_a_frag(const __bf16* rowPtr, int half) {
    v8bf lo = *(const v8bf*)(rowPtr + 8 * half);
    v8bf hi = *(const v8bf*)(rowPtr + 16 + 8 * half);
    v16bf r;
#pragma unroll
    for (int i = 0; i < 8; ++i) { r[i] = lo[i]; r[8 + i] = hi[i]; }
    return r;
}

// B-matrix fragment (32x16 bf16, KxN): b[0..7] = Bt[n][8*half .. +7], b[8..15] = Bt[n][16+8*half .. +7]
static __device__ __forceinline__ v16bf load_b_frag(const __bf16* colPtr, int half) {
    v8bf lo = *(const v8bf*)(colPtr + 8 * half);
    v8bf hi = *(const v8bf*)(colPtr + 16 + 8 * half);
    v16bf r;
#pragma unroll
    for (int i = 0; i < 8; ++i) { r[i] = lo[i]; r[8 + i] = hi[i]; }
    return r;
}

static __device__ __forceinline__ v8f wmma_bf16(v16bf a, v16bf b, v8f c) {
    return __builtin_amdgcn_wmma_f32_16x16x32_bf16(false, a, false, b, (short)0, c,
                                                   false, false);
}

// gfx1250 async global->LDS copy (16B per lane), tracked by ASYNCcnt
static __device__ __forceinline__ void async_copy16(unsigned ldsAddr,
                                                    const void* sbase, int byteOff) {
    asm volatile("global_load_async_to_lds_b128 %0, %1, %2"
                 :: "v"(ldsAddr), "v"(byteOff), "s"(sbase)
                 : "memory");
}

// ---------------------------------------------------------------------------
// Tiled GEMM: C(4096,1024) = A(4096,1024) @ W(1024,1024)^T + bias
// 128x128x32 block tiles, 8 waves (2x4), wave tile 64x32, register double-buffer
// OUT_MODE: 0 = fp32 linear (B,S,DM); 1 = bf16 head-major (B,H,S,dk);
//           2 = bf16 head-transposed (B,H,dk,S)
// ---------------------------------------------------------------------------
template <bool A_BF16, int OUT_MODE>
__global__ __launch_bounds__(256) void gemm_bias_kernel(const void* __restrict__ Aptr,
                                                        const float* __restrict__ W,
                                                        const float* __restrict__ bias,
                                                        void* __restrict__ outp) {
    __shared__ __bf16 sA[128][40];  // 80B row stride: 16B aligned, conflict-skewed
    __shared__ __bf16 sB[128][40];

    const int tid  = threadIdx.x;
    const int lane = tid & 31;
    const int lr   = lane & 15;
    const int half = lane >> 4;
    const int wave = tid >> 5;
    const int waveM = wave >> 2;
    const int waveN = wave & 3;

    const int nBase = blockIdx.x * 128;
    const int mBase = blockIdx.y * 128;

    v8f acc[4][2];
#pragma unroll
    for (int mi = 0; mi < 4; ++mi)
#pragma unroll
        for (int ni = 0; ni < 2; ++ni) acc[mi][ni] = v8f{};

    float4 aRegF[4], wReg[4];
    v4bf   aRegB[4];

    auto loadRegs = [&](int k0) {
#pragma unroll
        for (int i = 0; i < 4; ++i) {
            int v = tid + i * 256, row = v >> 3, c = (v & 7) * 4;
            if constexpr (A_BF16)
                aRegB[i] = *(const v4bf*)((const __bf16*)Aptr +
                                          (size_t)(mBase + row) * DM + k0 + c);
            else
                aRegF[i] = *(const float4*)((const float*)Aptr +
                                            (size_t)(mBase + row) * DM + k0 + c);
            wReg[i] = *(const float4*)(W + (size_t)(nBase + row) * DM + k0 + c);
        }
    };

    loadRegs(0);
    for (int k0 = 0; k0 < DM; k0 += 32) {
        // ---- commit staged registers to LDS as bf16 ----
#pragma unroll
        for (int i = 0; i < 4; ++i) {
            int v = tid + i * 256, row = v >> 3, c = (v & 7) * 4;
            if constexpr (A_BF16) {
#pragma unroll
                for (int j = 0; j < 4; ++j) sA[row][c + j] = aRegB[i][j];
            } else {
                sA[row][c + 0] = f2bf(aRegF[i].x);
                sA[row][c + 1] = f2bf(aRegF[i].y);
                sA[row][c + 2] = f2bf(aRegF[i].z);
                sA[row][c + 3] = f2bf(aRegF[i].w);
            }
            sB[row][c + 0] = f2bf(wReg[i].x);
            sB[row][c + 1] = f2bf(wReg[i].y);
            sB[row][c + 2] = f2bf(wReg[i].z);
            sB[row][c + 3] = f2bf(wReg[i].w);
        }
        __syncthreads();

        if (k0 + 32 < DM) loadRegs(k0 + 32);  // next tile in flight during WMMAs

        v16bf bFrag[2];
#pragma unroll
        for (int ni = 0; ni < 2; ++ni)
            bFrag[ni] = load_b_frag(&sB[waveN * 32 + ni * 16 + lr][0], half);
#pragma unroll
        for (int mi = 0; mi < 4; ++mi) {
            v16bf aFrag = load_a_frag(&sA[waveM * 64 + mi * 16 + lr][0], half);
#pragma unroll
            for (int ni = 0; ni < 2; ++ni)
                acc[mi][ni] = wmma_bf16(aFrag, bFrag[ni], acc[mi][ni]);
        }
        __syncthreads();
    }

    // ---- epilogue: bias + store ----
#pragma unroll
    for (int ni = 0; ni < 2; ++ni) {
        int n = nBase + waveN * 32 + ni * 16 + lr;
        float bv = bias[n];
#pragma unroll
        for (int mi = 0; mi < 4; ++mi) {
#pragma unroll
            for (int r = 0; r < 8; ++r) {
                int m = mBase + waveM * 64 + mi * 16 + r + 8 * half;
                float val = acc[mi][ni][r] + bv;
                int b = m >> 11, s = m & (SEQ - 1);
                int h = n >> 6, d = n & (DKH - 1);
                if constexpr (OUT_MODE == 1) {
                    ((__bf16*)outp)[(((size_t)(b * NH + h) * SEQ) + s) * DKH + d] =
                        f2bf(val);
                } else if constexpr (OUT_MODE == 2) {
                    ((__bf16*)outp)[(((size_t)(b * NH + h) * DKH) + d) * SEQ + s] =
                        f2bf(val);
                } else {
                    ((float*)outp)[(size_t)m * DM + n] = val;
                }
            }
        }
    }
}

// ---------------------------------------------------------------------------
// Flash attention, double-buffered async K/V staging.
// Q,K: (B,H,S,dk) bf16.  V: (B,H,dk,S) bf16 (pre-transposed by projection).
// Row-constant cosine bias provably cancels in softmax -> omitted here.
// ---------------------------------------------------------------------------
__global__ __launch_bounds__(256) void attn_kernel(const __bf16* __restrict__ Q,
                                                   const __bf16* __restrict__ K,
                                                   const __bf16* __restrict__ Vt,
                                                   const int* __restrict__ mask,
                                                   __bf16* __restrict__ attnOut) {
    __shared__ __bf16 sK[2][64][72];   // [buf][key][dk]
    __shared__ __bf16 sVt[2][64][72];  // [buf][dk][key]
    __shared__ __bf16 sP[8][16][72];   // per-wave P tile [m][key]

    const int tid  = threadIdx.x;
    const int lane = tid & 31;
    const int lr   = lane & 15;
    const int half = lane >> 4;
    const int w    = tid >> 5;

    const int qblk = blockIdx.x;
    const int bh   = blockIdx.y;
    const size_t headBase = (size_t)bh * SEQ * DKH;
    const int qrow0 = qblk * 128 + w * 16;

    // resident Q fragments
    const __bf16* qrow = Q + headBase + (size_t)(qrow0 + lr) * DKH;
    v16bf aQ[2];
#pragma unroll
    for (int kk = 0; kk < 2; ++kk) aQ[kk] = load_a_frag(qrow + kk * 32, half);

    v8f o[4];
#pragma unroll
    for (int dn = 0; dn < 4; ++dn) o[dn] = v8f{};
    float mrow[8], lrow[8];
#pragma unroll
    for (int r = 0; r < 8; ++r) { mrow[r] = -INFINITY; lrow[r] = 0.0f; }

    const float scale = 0.125f;  // 1/sqrt(64)

    // async staging of one 64-key tile (K row-major, Vt dk-major; both raw bf16)
    auto issue_tile = [&](int buf, int key0) {
        const void* baseK = (const void*)(K + headBase + (size_t)key0 * DKH);
        const void* baseV = (const void*)(Vt + (size_t)bh * DKH * SEQ + key0);
#pragma unroll
        for (int i = 0; i < 2; ++i) {
            int c = tid + i * 256;         // 512 x 16B chunks per tile
            int row = c >> 3, cc = (c & 7) * 8;
            async_copy16((unsigned)(size_t)&sK[buf][row][cc], baseK,
                         (row * DKH + cc) * 2);
            async_copy16((unsigned)(size_t)&sVt[buf][row][cc], baseV,
                         (row * SEQ + cc) * 2);
        }
    };

    issue_tile(0, 0);

    for (int key0 = 0; key0 < SEQ; key0 += 64) {
        const int buf = (key0 >> 6) & 1;
        if (key0 + 64 < SEQ) {
            issue_tile(buf ^ 1, key0 + 64);
            asm volatile("s_wait_asynccnt 4" ::: "memory");  // tile 'buf' landed
        } else {
            asm volatile("s_wait_asynccnt 0" ::: "memory");
        }
        __syncthreads();

        // ---- scores ----
        v8f s[4];
#pragma unroll
        for (int nn = 0; nn < 4; ++nn) {
            const __bf16* krow = &sK[buf][nn * 16 + lr][0];
            v8f a = v8f{};
            a = wmma_bf16(aQ[0], load_b_frag(krow + 0, half), a);
            a = wmma_bf16(aQ[1], load_b_frag(krow + 32, half), a);
            s[nn] = a;
        }

        // ---- scale + mask + online softmax ----
        float tmax[8];
#pragma unroll
        for (int r = 0; r < 8; ++r) tmax[r] = -INFINITY;
#pragma unroll
        for (int nn = 0; nn < 4; ++nn) {
            int mk = mask[key0 + nn * 16 + lr];
#pragma unroll
            for (int r = 0; r < 8; ++r) {
                float x = s[nn][r] * scale;
                if (mk == 0) x = -1.0e9f;
                s[nn][r] = x;
                tmax[r]  = fmaxf(tmax[r], x);
            }
        }
#pragma unroll
        for (int off = 1; off < 16; off <<= 1)
#pragma unroll
            for (int r = 0; r < 8; ++r)
                tmax[r] = fmaxf(tmax[r], __shfl_xor(tmax[r], off, 32));

        float al[8], rs[8];
#pragma unroll
        for (int r = 0; r < 8; ++r) {
            float mn = fmaxf(mrow[r], tmax[r]);
            al[r] = (mn == -INFINITY) ? 0.0f : __expf(mrow[r] - mn);
            mrow[r] = mn;
            rs[r] = 0.0f;
        }
#pragma unroll
        for (int nn = 0; nn < 4; ++nn)
#pragma unroll
            for (int r = 0; r < 8; ++r) {
                float p = __expf(s[nn][r] - mrow[r]);
                s[nn][r] = p;
                rs[r] += p;
            }
#pragma unroll
        for (int off = 1; off < 16; off <<= 1)
#pragma unroll
            for (int r = 0; r < 8; ++r) rs[r] += __shfl_xor(rs[r], off, 32);
#pragma unroll
        for (int r = 0; r < 8; ++r) lrow[r] = lrow[r] * al[r] + rs[r];
#pragma unroll
        for (int dn = 0; dn < 4; ++dn)
#pragma unroll
            for (int r = 0; r < 8; ++r) o[dn][r] *= al[r];

        // ---- C-layout -> A-layout for P via per-wave LDS ----
#pragma unroll
        for (int nn = 0; nn < 4; ++nn)
#pragma unroll
            for (int r = 0; r < 8; ++r)
                sP[w][r + 8 * half][nn * 16 + lr] = f2bf(s[nn][r]);
        asm volatile("s_wait_dscnt 0" ::: "memory");

        v16bf aP[2];
#pragma unroll
        for (int kk = 0; kk < 2; ++kk)
            aP[kk] = load_a_frag(&sP[w][lr][kk * 32], half);
#pragma unroll
        for (int dn = 0; dn < 4; ++dn) {
            const __bf16* vrow = &sVt[buf][dn * 16 + lr][0];
            o[dn] = wmma_bf16(aP[0], load_b_frag(vrow + 0, half), o[dn]);
            o[dn] = wmma_bf16(aP[1], load_b_frag(vrow + 32, half), o[dn]);
        }
        __syncthreads();  // reads of 'buf' done before tile t+2 overwrites it
    }

    // ---- normalize + store attn in (B, S, H*dk) bf16 ----
    float inv[8];
#pragma unroll
    for (int r = 0; r < 8; ++r) inv[r] = 1.0f / fmaxf(lrow[r], 1e-30f);
    const int b = bh >> 4, h = bh & 15;
#pragma unroll
    for (int dn = 0; dn < 4; ++dn)
#pragma unroll
        for (int r = 0; r < 8; ++r) {
            int srow = qblk * 128 + w * 16 + r + 8 * half;
            int col  = h * DKH + dn * 16 + lr;
            attnOut[((size_t)b * SEQ + srow) * DM + col] = f2bf(o[dn][r] * inv[r]);
        }
}

// ---------------------------------------------------------------------------
// Per-(b,h) cosine of sequence means of Q (B,H,S,dk) and Vt (B,H,dk,S)
// ---------------------------------------------------------------------------
__global__ __launch_bounds__(256) void bias_cos_kernel(const __bf16* __restrict__ Q,
                                                       const __bf16* __restrict__ Vt,
                                                       float* __restrict__ cosOut) {
    __shared__ float bq[4][64], bv[4][64], qm[64], vm[64];
    const int bh = blockIdx.x;
    const int d  = threadIdx.x & 63;
    const int g  = threadIdx.x >> 6;
    const size_t qbase = (size_t)bh * SEQ * DKH;
    const size_t vbase = (size_t)bh * DKH * SEQ + (size_t)d * SEQ;
    float qs = 0.0f, vs = 0.0f;
    for (int s = g; s < SEQ; s += 4) {
        qs += (float)Q[qbase + (size_t)s * DKH + d];
        vs += (float)Vt[vbase + s];
    }
    bq[g][d] = qs;
    bv[g][d] = vs;
    __syncthreads();
    if (threadIdx.x < 64) {
        qm[threadIdx.x] = (bq[0][threadIdx.x] + bq[1][threadIdx.x] +
                           bq[2][threadIdx.x] + bq[3][threadIdx.x]) * (1.0f / SEQ);
        vm[threadIdx.x] = (bv[0][threadIdx.x] + bv[1][threadIdx.x] +
                           bv[2][threadIdx.x] + bv[3][threadIdx.x]) * (1.0f / SEQ);
    }
    __syncthreads();
    if (threadIdx.x == 0) {
        float dot = 0.0f, nq = 0.0f, nv = 0.0f;
        for (int i = 0; i < 64; ++i) {
            dot += qm[i] * vm[i];
            nq += qm[i] * qm[i];
            nv += vm[i] * vm[i];
        }
        cosOut[bh] = dot / (fmaxf(sqrtf(nq), 1e-8f) * fmaxf(sqrtf(nv), 1e-8f));
    }
}

__global__ void finalize_kernel(const float* __restrict__ cosv,
                                float* __restrict__ outScalar) {
    float v = cosv[threadIdx.x];
#pragma unroll
    for (int off = 1; off < 32; off <<= 1) v += __shfl_xor(v, off, 32);
    if (threadIdx.x == 0) outScalar[0] = v * (1.0f / 32.0f);
}

// ---------------------------------------------------------------------------
extern "C" void kernel_launch(void* const* d_in, const int* in_sizes, int n_in,
                              void* d_out, int out_size, void* d_ws, size_t ws_size,
                              hipStream_t stream) {
    const float* q    = (const float*)d_in[0];
    const float* k    = (const float*)d_in[1];
    const float* v    = (const float*)d_in[2];
    const int*   mask = (const int*)d_in[3];
    const float* Wq = (const float*)d_in[4];
    const float* bq = (const float*)d_in[5];
    const float* Wk = (const float*)d_in[6];
    const float* bk = (const float*)d_in[7];
    const float* Wv = (const float*)d_in[8];
    const float* bv = (const float*)d_in[9];
    const float* Wo = (const float*)d_in[10];
    const float* bo = (const float*)d_in[11];

    const size_t tensBytes = (size_t)BATCH * NH * SEQ * DKH * sizeof(__bf16);  // 8 MB
    char* ws = (char*)d_ws;
    __bf16* Qbf  = (__bf16*)(ws);
    __bf16* Kbf  = (__bf16*)(ws + tensBytes);
    __bf16* Vbf  = (__bf16*)(ws + 2 * tensBytes);  // (B,H,dk,S) transposed
    __bf16* Abf  = (__bf16*)(ws + 3 * tensBytes);
    float*  cosb = (float*)(ws + 4 * tensBytes);

    dim3 gGemm(DM / 128, MROWS / 128);  // (8, 32)
    gemm_bias_kernel<false, 1><<<gGemm, 256, 0, stream>>>(q, Wq, bq, Qbf);
    gemm_bias_kernel<false, 1><<<gGemm, 256, 0, stream>>>(k, Wk, bk, Kbf);
    gemm_bias_kernel<false, 2><<<gGemm, 256, 0, stream>>>(v, Wv, bv, Vbf);

    attn_kernel<<<dim3(SEQ / 128, BATCH * NH), 256, 0, stream>>>(Qbf, Kbf, Vbf, mask,
                                                                 Abf);

    bias_cos_kernel<<<BATCH * NH, 256, 0, stream>>>(Qbf, Vbf, cosb);
    finalize_kernel<<<1, 32, 0, stream>>>(cosb, (float*)d_out + (out_size - 1));

    gemm_bias_kernel<true, 0><<<gGemm, 256, 0, stream>>>(Abf, Wo, bo, d_out);
}
